// FusingNet_11123965296814
// MI455X (gfx1250) — matrix operand order repaired and checked by reading, not compiled, not run on hardware
//
#include <hip/hip_runtime.h>

// ---------------------------------------------------------------------------
// FusingNet (kNN-MOS fusion) for MI455X / gfx1250, wave32 + WMMA bf16.
// v2: 32 query rows per block (2 M-tiles per wave) -> each B fragment feeds
//     2 WMMAs, halving L2 key traffic for the distance GEMM.
// ---------------------------------------------------------------------------

typedef __bf16 bf16_t;
typedef bf16_t v16bf __attribute__((ext_vector_type(16)));
typedef float  v8f   __attribute__((ext_vector_type(8)));
typedef unsigned int v4u __attribute__((ext_vector_type(4)));
typedef unsigned int v8u __attribute__((ext_vector_type(8)));
typedef unsigned short ushort_t;

#define BQ_TOTAL 2048
#define NKEYS    50000
#define DIM      768
#define MAXK     16
#define TOPK     8
#define NBINS    16
#define NSLAB    8
#define SLABC    6400   // 25 groups of 256 per slab; 8*6400 = 51200 >= 50000
#define GROUP    256
#define MROWS    32     // query rows per block (2 WMMA M-tiles)
#define BIGD     1e30f

__device__ __forceinline__ ushort_t f2bf_rne(float f) {
    unsigned int u = __float_as_uint(f);
    unsigned int r = u + 0x7FFFu + ((u >> 16) & 1u);
    return (ushort_t)(r >> 16);
}

// ---------------------------------------------------------------------------
// Kernel 1: fp32 -> bf16 conversion + row squared norms for Q and K.
// One 256-thread block per row (DIM = 768 = 3*256).
// ---------------------------------------------------------------------------
__global__ __launch_bounds__(256)
void prep_kernel(const float* __restrict__ qsrc, const float* __restrict__ ksrc,
                 ushort_t* __restrict__ qb, ushort_t* __restrict__ kb,
                 float* __restrict__ q2, float* __restrict__ k2) {
    __shared__ float red[256];
    int row = blockIdx.x;
    int tid = threadIdx.x;

    const float* src;
    ushort_t* dst;
    float* outsq;
    if (row < BQ_TOTAL) {
        src = qsrc + (size_t)row * DIM;
        dst = qb + (size_t)row * DIM;
        outsq = q2 + row;
    } else {
        int rr = row - BQ_TOTAL;
        src = ksrc + (size_t)rr * DIM;
        dst = kb + (size_t)rr * DIM;
        outsq = k2 + rr;
    }

    float s = 0.0f;
#pragma unroll
    for (int k = 0; k < 3; ++k) {
        int i = tid + k * 256;
        float v = src[i];
        dst[i] = f2bf_rne(v);
        s += v * v;
    }
    red[tid] = s;
    __syncthreads();
    for (int off = 128; off > 0; off >>= 1) {
        if (tid < off) red[tid] += red[tid + off];
        __syncthreads();
    }
    if (tid == 0) *outsq = red[0];
}

// ---------------------------------------------------------------------------
// Kernel 2: fused masked-distance GEMM + per-slab top-16.
// Grid: (2048/32 query tiles, NSLAB).  Block: 128 threads = 4 waves.
// Each wave: 2 M-tiles x 4 N-tiles (32 rows x 64 cols); block: 32 x 256/iter.
// ---------------------------------------------------------------------------
__global__ __launch_bounds__(128)
void knn_kernel(const ushort_t* __restrict__ qb, const ushort_t* __restrict__ kb,
                const float* __restrict__ q2, const float* __restrict__ k2,
                const int* __restrict__ syslist, const int* __restrict__ key_sys,
                float* __restrict__ candD, int* __restrict__ candI) {
    __shared__ float sc[MROWS][GROUP];
    __shared__ float topd[MROWS][MAXK];
    __shared__ int   topi[MROWS][MAXK];
    __shared__ float lq2[MROWS];
    __shared__ int   lsys[MROWS];

    const int tid  = threadIdx.x;
    const int wave = tid >> 5;
    const int lane = tid & 31;
    const int r    = lane & 15;   // row-in-tile (A) / column-in-tile (B,C)
    const int g    = lane >> 4;   // half-wave selector
    const int qtile = blockIdx.x;
    const int slab  = blockIdx.y;

    if (tid < MROWS) {
        lq2[tid]  = q2[qtile * MROWS + tid];
        lsys[tid] = syslist[qtile * MROWS + tid];
#pragma unroll
        for (int j = 0; j < MAXK; ++j) { topd[tid][j] = 3e37f; topi[tid][j] = 0; }
    }
    __syncthreads();

    const ushort_t* qrow0 = qb + (size_t)(qtile * MROWS + r) * DIM;
    const ushort_t* qrow1 = qb + (size_t)(qtile * MROWS + 16 + r) * DIM;

    for (int grp = 0; grp < SLABC / GROUP; ++grp) {
        const int base = slab * SLABC + grp * GROUP;

        v8f acc[2][4];
#pragma unroll
        for (int m = 0; m < 2; ++m)
#pragma unroll
            for (int t = 0; t < 4; ++t)
                acc[m][t] = v8f{0.f,0.f,0.f,0.f,0.f,0.f,0.f,0.f};

        // B fragment row pointers (clamped for the N tail; masked later)
        const ushort_t* krow[4];
#pragma unroll
        for (int t = 0; t < 4; ++t) {
            int col = base + (wave * 4 + t) * 16 + r;
            int cc = col < NKEYS ? col : (NKEYS - 1);
            krow[t] = kb + (size_t)cc * DIM + g * 16;
        }

        for (int kbase = 0; kbase < DIM; kbase += 32) {
            // A fragments: lane holds row r, K = {g*16..g*16+15}
            v4u a0lo = *(const v4u*)(qrow0 + kbase + g * 16);
            v4u a0hi = *(const v4u*)(qrow0 + kbase + 8 + g * 16);
            v16bf A0 = __builtin_bit_cast(
                v16bf, __builtin_shufflevector(a0lo, a0hi, 0, 1, 2, 3, 4, 5, 6, 7));
            v4u a1lo = *(const v4u*)(qrow1 + kbase + g * 16);
            v4u a1hi = *(const v4u*)(qrow1 + kbase + 8 + g * 16);
            v16bf A1 = __builtin_bit_cast(
                v16bf, __builtin_shufflevector(a1lo, a1hi, 0, 1, 2, 3, 4, 5, 6, 7));
#pragma unroll
            for (int t = 0; t < 4; ++t) {
                // B fragment: lane holds one column, 16 contiguous K values;
                // reused by both M-tiles (halves L2 key traffic).
                v8u bw  = *(const v8u*)(krow[t] + kbase);
                v16bf Bf = __builtin_bit_cast(v16bf, bw);
                acc[0][t] = __builtin_amdgcn_wmma_f32_16x16x32_bf16(
                    false, A0, false, Bf, (short)0, acc[0][t], false, false);
                acc[1][t] = __builtin_amdgcn_wmma_f32_16x16x32_bf16(
                    false, A1, false, Bf, (short)0, acc[1][t], false, false);
            }
        }

        // Epilogue: d2 = q2 + k2 - 2*dot, mask same-system + tail, to LDS.
#pragma unroll
        for (int t = 0; t < 4; ++t) {
            int col = base + (wave * 4 + t) * 16 + r;
            bool in = (col < NKEYS);
            float k2c = in ? k2[col] : 0.0f;
            int   ks  = in ? key_sys[col] : -1;
#pragma unroll
            for (int m = 0; m < 2; ++m) {
#pragma unroll
                for (int j = 0; j < 8; ++j) {
                    int row = m * 16 + j + g * 8;
                    float d2 = lq2[row] + k2c - 2.0f * acc[m][t][j];
                    bool bad = (!in) || (ks == lsys[row]);
                    sc[row][(wave * 4 + t) * 16 + r] = bad ? BIGD : d2;
                }
            }
        }
        __syncthreads();

        // 32 threads: threshold-gated insertion into sorted top-16.
        if (tid < MROWS) {
            const int q = tid;
            float wd = topd[q][MAXK - 1];
            for (int c = 0; c < GROUP; ++c) {
                float d = sc[q][c];
                if (d < wd) {
                    int col = base + c;
                    int k = MAXK - 1;
                    while (k > 0 && topd[q][k - 1] > d) {
                        topd[q][k] = topd[q][k - 1];
                        topi[q][k] = topi[q][k - 1];
                        --k;
                    }
                    topd[q][k] = d;
                    topi[q][k] = col;
                    wd = topd[q][MAXK - 1];
                }
            }
        }
        __syncthreads();
    }

    if (tid < MROWS) {
        size_t o = ((size_t)(qtile * MROWS + tid) * NSLAB + slab) * MAXK;
#pragma unroll
        for (int j = 0; j < MAXK; ++j) { candD[o + j] = topd[tid][j]; candI[o + j] = topi[tid][j]; }
    }
}

// ---------------------------------------------------------------------------
// Kernel 3: merge per-slab candidates -> global top-16, then the fusion head.
// One wave (32 threads) per query; hidden layers are width 32 = 1 unit/lane.
// ---------------------------------------------------------------------------
__global__ __launch_bounds__(32)
void head_kernel(const float* __restrict__ candD, const int* __restrict__ candI,
                 const float* __restrict__ vals,
                 const float* __restrict__ wscore, const float* __restrict__ wprob,
                 const float* __restrict__ W1k, const float* __restrict__ b1k,
                 const float* __restrict__ W2k, const float* __restrict__ b2k,
                 const float* __restrict__ W1l, const float* __restrict__ b1l,
                 const float* __restrict__ W2l, const float* __restrict__ b2l,
                 const float* __restrict__ W1w, const float* __restrict__ b1w,
                 const float* __restrict__ W2w, const float* __restrict__ b2w,
                 float* __restrict__ out) {
    __shared__ float bd[MAXK];
    __shared__ int   bi[MAXK];
    __shared__ float x[MAXK];     // knn_dists (negated d2)
    __shared__ float scv[MAXK];   // knn_scores
    __shared__ float h[32];
    __shared__ float lg[MAXK];
    __shared__ float wi[MAXK + TOPK + 2];
    __shared__ float scl[4];      // [knn_result, knn_lambda, wav_lambda, spare]

    const int q = blockIdx.x;
    const int lane = threadIdx.x;

    // --- merge 8 sorted 16-lists -> global top-16 (serial on lane 0) ---
    if (lane == 0) {
#pragma unroll
        for (int j = 0; j < MAXK; ++j) { bd[j] = 3e37f; bi[j] = 0; }
        size_t o = (size_t)q * NSLAB * MAXK;
        for (int c = 0; c < NSLAB * MAXK; ++c) {
            float d = candD[o + c];
            if (d < bd[MAXK - 1]) {
                int id = candI[o + c];
                int k = MAXK - 1;
                while (k > 0 && bd[k - 1] > d) { bd[k] = bd[k - 1]; bi[k] = bi[k - 1]; --k; }
                bd[k] = d; bi[k] = id;
            }
        }
    }
    __syncthreads();

    if (lane < MAXK) {
        x[lane]   = -bd[lane];          // top_k(-d2): descending neg-dist
        scv[lane] = vals[bi[lane]];
    }
    __syncthreads();

    // --- k_prob MLP: 16 -> 32 (tanh) -> 16 -> softmax ---
    {
        float s = b1k[lane];
#pragma unroll
        for (int i = 0; i < MAXK; ++i) s += x[i] * W1k[i * 32 + lane];
        h[lane] = tanhf(s);
    }
    __syncthreads();
    if (lane < MAXK) {
        float s = b2k[lane];
#pragma unroll
        for (int j = 0; j < 32; ++j) s += h[j] * W2k[j * MAXK + lane];
        lg[lane] = s;
    }
    __syncthreads();
    if (lane == 0) {
        float m = lg[0];
        for (int i = 1; i < MAXK; ++i) m = fmaxf(m, lg[i]);
        float den = 0.0f, num = 0.0f;
        for (int i = 0; i < MAXK; ++i) {
            float e = __expf(lg[i] - m);
            den += e;
            num += e * scv[i];
        }
        scl[0] = num / den;  // knn_result
    }
    __syncthreads();

    const float knn_result = scl[0];
    const float wscore_q = wscore[q];
    const float* prow = wprob + (size_t)q * NBINS;

    // --- wav_in = [prob[wav_idx], prob[knn_idx], top8(prob), knn_dists] ---
    if (lane == 0) {
        float vw = fminf(fmaxf(wscore_q * 4.0f - 4.0f, 0.0f), 15.0f);
        float vk = fminf(fmaxf(knn_result * 4.0f - 4.0f, 0.0f), 15.0f);
        wi[0] = prow[(int)vw];
        wi[1] = prow[(int)vk];
        float tmp[NBINS];
#pragma unroll
        for (int i = 0; i < NBINS; ++i) tmp[i] = prow[i];
        for (int s = 0; s < TOPK; ++s) {          // partial selection sort, desc
            int best = s;
            for (int i = s + 1; i < NBINS; ++i) if (tmp[i] > tmp[best]) best = i;
            float t = tmp[s]; tmp[s] = tmp[best]; tmp[best] = t;
            wi[2 + s] = tmp[s];
        }
    }
    if (lane < MAXK) wi[2 + TOPK + lane] = x[lane];
    __syncthreads();

    // --- knn_lambda MLP: 16 -> 32 (tanh) -> 1 ---
    {
        float s = b1l[lane];
#pragma unroll
        for (int i = 0; i < MAXK; ++i) s += x[i] * W1l[i * 32 + lane];
        h[lane] = tanhf(s);
    }
    __syncthreads();
    if (lane == 0) {
        float s = b2l[0];
        for (int j = 0; j < 32; ++j) s += h[j] * W2l[j];
        scl[1] = s;
    }
    __syncthreads();

    // --- wav_lambda MLP: 26 -> 32 (tanh) -> 1 ---
    {
        float s = b1w[lane];
#pragma unroll
        for (int i = 0; i < MAXK + TOPK + 2; ++i) s += wi[i] * W1w[i * 32 + lane];
        h[lane] = tanhf(s);
    }
    __syncthreads();
    if (lane == 0) {
        float s = b2w[0];
        for (int j = 0; j < 32; ++j) s += h[j] * W2w[j];
        float kl = scl[1];
        float m = fmaxf(kl, s);
        float e0 = __expf(kl - m), e1 = __expf(s - m);
        float inv = 1.0f / (e0 + e1);
        out[q] = (e0 * inv) * knn_result + (e1 * inv) * wscore_q;
    }
}

// ---------------------------------------------------------------------------
// Launch
// ---------------------------------------------------------------------------
static inline size_t align256(size_t x) { return (x + 255) & ~(size_t)255; }

extern "C" void kernel_launch(void* const* d_in, const int* in_sizes, int n_in,
                              void* d_out, int out_size, void* d_ws, size_t ws_size,
                              hipStream_t stream) {
    (void)in_sizes; (void)n_in; (void)out_size; (void)ws_size;

    const float* input_future = (const float*)d_in[0];
    const float* wscore       = (const float*)d_in[1];
    const float* wprob        = (const float*)d_in[2];
    const int*   syslist      = (const int*)d_in[3];
    const float* keys         = (const float*)d_in[4];
    const float* vals         = (const float*)d_in[5];
    const int*   key_sys      = (const int*)d_in[6];
    const float* W1k = (const float*)d_in[7];
    const float* b1k = (const float*)d_in[8];
    const float* W2k = (const float*)d_in[9];
    const float* b2k = (const float*)d_in[10];
    const float* W1l = (const float*)d_in[11];
    const float* b1l = (const float*)d_in[12];
    const float* W2l = (const float*)d_in[13];
    const float* b2l = (const float*)d_in[14];
    const float* W1w = (const float*)d_in[15];
    const float* b1w = (const float*)d_in[16];
    const float* W2w = (const float*)d_in[17];
    const float* b2w = (const float*)d_in[18];
    float* out = (float*)d_out;

    char* ws = (char*)d_ws;
    size_t o = 0;
    ushort_t* qb = (ushort_t*)(ws + o); o = align256(o + (size_t)BQ_TOTAL * DIM * 2);
    ushort_t* kb = (ushort_t*)(ws + o); o = align256(o + (size_t)NKEYS * DIM * 2);
    float* q2 = (float*)(ws + o);       o = align256(o + (size_t)BQ_TOTAL * 4);
    float* k2 = (float*)(ws + o);       o = align256(o + (size_t)NKEYS * 4);
    float* candD = (float*)(ws + o);    o = align256(o + (size_t)BQ_TOTAL * NSLAB * MAXK * 4);
    int*   candI = (int*)(ws + o);      o = align256(o + (size_t)BQ_TOTAL * NSLAB * MAXK * 4);

    prep_kernel<<<BQ_TOTAL + NKEYS, 256, 0, stream>>>(input_future, keys, qb, kb, q2, k2);

    dim3 grid(BQ_TOTAL / MROWS, NSLAB);
    knn_kernel<<<grid, 128, 0, stream>>>(qb, kb, q2, k2, syslist, key_sys, candD, candI);

    head_kernel<<<BQ_TOTAL, 32, 0, stream>>>(candD, candI, vals, wscore, wprob,
                                             W1k, b1k, W2k, b2k,
                                             W1l, b1l, W2l, b2l,
                                             W1w, b1w, W2w, b2w, out);
}
